// OptimizedCustomLSTM_10471130268441
// MI455X (gfx1250) — matrix operand order repaired and checked by reading, not compiled, not run on hardware
//
#include <hip/hip_runtime.h>

// ---------------------------------------------------------------------------
// LSTM (SEQ=512, BATCH=64, INPUT=1024, HIDDEN=1024) for gfx1250 (MI455X).
// Phase 1: x_gates = x @ Wx^T + bx  via bf16 WMMA (f32 accumulate), bf16 out.
// Phase 2: 512 sequential steps, each gates = xg_t + h @ Wh^T + bh via WMMA,
//          then fused sigmoid/tanh cell update.
// Data movement: async global->LDS copies (ASYNCcnt) for h / Wx tiles and a
// TDM tensor_load_to_lds (TENSORcnt) for the strided Wh tile, when available.
// ---------------------------------------------------------------------------

#define SEQ    512
#define BATCH  64
#define INPUT  1024
#define HIDDEN 1024
#define GATES  (4 * HIDDEN)

typedef __attribute__((ext_vector_type(16))) __bf16 v16bf;
typedef __attribute__((ext_vector_type(8)))  __bf16 v8bf;
typedef __attribute__((ext_vector_type(8)))  float  v8f;

typedef unsigned int u32x4 __attribute__((ext_vector_type(4)));
typedef int          i32x8 __attribute__((ext_vector_type(8)));
typedef int          i32x4 __attribute__((ext_vector_type(4)));

#define AS1 __attribute__((address_space(1)))
#define AS3 __attribute__((address_space(3)))

#if defined(__has_builtin)
#  if __has_builtin(__builtin_amdgcn_global_load_async_to_lds_b128) && \
      __has_builtin(__builtin_amdgcn_s_wait_asynccnt)
#    define USE_ASYNC_LDS 1
#  endif
#  if __has_builtin(__builtin_amdgcn_tensor_load_to_lds) && \
      __has_builtin(__builtin_amdgcn_s_wait_tensorcnt)
#    define USE_TDM 1
#  endif
#endif
#ifndef USE_ASYNC_LDS
#  define USE_ASYNC_LDS 0
#endif
#ifndef USE_TDM
#  define USE_TDM 0
#endif

// 16-byte global -> LDS copy: async (no VGPR data, ASYNCcnt) when available.
// b128 variant is typed as <4 x i32> pointers in AS1/AS3 (per hipcc diag).
__device__ __forceinline__ void copy16_g2l(const __bf16* g, __bf16* l) {
#if USE_ASYNC_LDS
    __builtin_amdgcn_global_load_async_to_lds_b128(
        (AS1 i32x4*)g, (AS3 i32x4*)l, 0, 0);
#else
    *(v8bf*)l = *(const v8bf*)g;
#endif
}

__device__ __forceinline__ void copy_fence_pre_barrier() {
#if USE_ASYNC_LDS
    __builtin_amdgcn_s_wait_asynccnt(0);
#endif
}

// ---------------------------------------------------------------------------
// A-matrix fragment (16x32 bf16) from an LDS tile laid out [rows][32].
// Lane L (<16):  row L,    K = {0..7, 16..23}
// Lane L (>=16): row L-16, K = {8..15, 24..31}
// ---------------------------------------------------------------------------
__device__ __forceinline__ v16bf load_a_frag_lds(const __bf16* base, int lane) {
    int row  = lane & 15;
    int koff = (lane >> 4) ? 8 : 0;
    const __bf16* p = base + row * 32 + koff;
    v8bf lo = *(const v8bf*)(p);        // K 0..7   (or 8..15)
    v8bf hi = *(const v8bf*)(p + 16);   // K 16..23 (or 24..31)
    v16bf a;
#pragma unroll
    for (int i = 0; i < 8; ++i) { a[i] = lo[i]; a[8 + i] = hi[i]; }
    return a;
}

// ---------------------------------------------------------------------------
// B-matrix fragment (32x16 bf16) for C[m][n] = sum_k A[m][k] * W[n][k].
// W row-major [Ncols][ldk].  Lane n (<16): col n, K = {0..7, 16..23}.
// Lane n (>=16): col n-16, K = {8..15, 24..31}.  Two 16B loads per lane.
// ---------------------------------------------------------------------------
__device__ __forceinline__ v16bf load_b_frag_global(const __bf16* W, int n0,
                                                    int kk, int ldk, int lane) {
    int col  = lane & 15;
    int koff = (lane >> 4) ? 8 : 0;
    const __bf16* p = W + (size_t)(n0 + col) * ldk + kk + koff;
    v8bf lo = *(const v8bf*)(p);
    v8bf hi = *(const v8bf*)(p + 16);
    v16bf b;
#pragma unroll
    for (int i = 0; i < 8; ++i) { b[i] = lo[i]; b[8 + i] = hi[i]; }
    return b;
}

__device__ __forceinline__ v16bf load_b_frag_lds(const __bf16* base, int lane) {
    int col  = lane & 15;
    int koff = (lane >> 4) ? 8 : 0;
    const __bf16* p = base + col * 32 + koff;
    v8bf lo = *(const v8bf*)(p);
    v8bf hi = *(const v8bf*)(p + 16);
    v16bf b;
#pragma unroll
    for (int i = 0; i < 8; ++i) { b[i] = lo[i]; b[8 + i] = hi[i]; }
    return b;
}

__device__ __forceinline__ v8f wmma_bf16(v16bf a, v16bf b, v8f c) {
    return __builtin_amdgcn_wmma_f32_16x16x32_bf16(
        /*neg_a=*/false, a, /*neg_b=*/false, b,
        /*c_mod=*/(short)0, c, /*reuse_a=*/false, /*reuse_b=*/false);
}

__device__ __forceinline__ float fsigmoid(float x) {
    return 1.0f / (1.0f + __expf(-x));
}

#if USE_TDM
// ---------------------------------------------------------------------------
// TDM: DMA the per-block Wh working set for one k-chunk into LDS.
// Tensor view of Wh (row-major [4H][H], 2B elems): 3-D [k=H][row=H][gate=4]
//   dim0 stride = H (next row), dim1 stride = H*H (next gate).
// Tile: [k=32][rows=16][gates=4] starting at (kk, jblk*16, 0)  -> 4 KB,
// landing contiguously in LDS as [gate][row][32] (gate-major rows).
// D# packing per CDNA5 ISA 8.3-8.6; 6-arg builtin form on this toolchain
// (trailing i32x8 uncharacterized -> zeros, cpol=0).
// ---------------------------------------------------------------------------
__device__ __forceinline__ void tdm_load_wh(const __bf16* Wh_bf,
                                            __bf16* lds_dst,
                                            int jblk, int kk) {
    unsigned long long gaddr =
        (unsigned long long)(uintptr_t)(Wh_bf + (size_t)(jblk * 16) * HIDDEN + kk);
    unsigned lds_addr = (unsigned)(uintptr_t)(AS3 void*)lds_dst;

    u32x4 g0;
    g0[0] = 1u;                                       // count=1, user mode
    g0[1] = lds_addr;                                 // LDS byte address
    g0[2] = (unsigned)(gaddr & 0xFFFFFFFFull);        // global addr lo
    g0[3] = (unsigned)((gaddr >> 32) & 0x01FFFFFFull) // global addr hi (57b)
            | (2u << 30);                             // type = 2 ("image")

    const unsigned long long data_size = 1;           // 2-byte elements
    const unsigned long long tdim0 = HIDDEN;          // k extent
    const unsigned long long tdim1 = HIDDEN;          // rows per gate
    const unsigned long long til0 = 32, til1 = 16, til2 = 4;
    const unsigned long long str0 = HIDDEN;                           // row
    const unsigned long long str1 = (unsigned long long)HIDDEN * HIDDEN; // gate

    unsigned long long q0 = (data_size << 16) | ((tdim0 & 0xFFFFull) << 48);
    unsigned long long q1 = (tdim0 >> 16) | (tdim1 << 16) | (til0 << 48);
    unsigned long long q2 = til1 | (til2 << 16) | ((str0 & 0xFFFFFFFFull) << 32);
    unsigned long long q3 = (str0 >> 32) | (str1 << 16);

    i32x8 g1;
    g1[0] = (int)(q0 & 0xFFFFFFFFull); g1[1] = (int)(q0 >> 32);
    g1[2] = (int)(q1 & 0xFFFFFFFFull); g1[3] = (int)(q1 >> 32);
    g1[4] = (int)(q2 & 0xFFFFFFFFull); g1[5] = (int)(q2 >> 32);
    g1[6] = (int)(q3 & 0xFFFFFFFFull); g1[7] = (int)(q3 >> 32);

    i32x4 g2;
    g2[0] = 4;    // tensor_dim2 (gates)
    g2[1] = 0;    // tensor_dim3
    g2[2] = 0;    // tensor_dim2_stride lo (unused)
    g2[3] = 0;    // tensor_dim2_stride hi | tile_dim3=0

    i32x4 g3 = {0, 0, 0, 0};
    i32x8 g4 = {0, 0, 0, 0, 0, 0, 0, 0};   // trailing group (6-arg form)

    __builtin_amdgcn_tensor_load_to_lds(g0, g1, g2, g3, g4, 0);
}
#endif // USE_TDM

// ---------------------------------------------------------------------------
// Kernel: convert Wx and Wh (f32) to bf16 in workspace.
// ---------------------------------------------------------------------------
__global__ void lstm_convert_weights(const float* __restrict__ Wx,
                                     const float* __restrict__ Wh,
                                     __bf16* __restrict__ Wx_bf,
                                     __bf16* __restrict__ Wh_bf) {
    const size_t N = (size_t)GATES * INPUT;   // 4M per matrix
    size_t i = (size_t)blockIdx.x * blockDim.x + threadIdx.x;
    if (i < N)           Wx_bf[i]     = (__bf16)Wx[i];
    else if (i < 2 * N)  Wh_bf[i - N] = (__bf16)Wh[i - N];
}

// ---------------------------------------------------------------------------
// Kernel: zero h0 (both double-buffer slots) and c0.
// ---------------------------------------------------------------------------
__global__ void lstm_init_state(__bf16* __restrict__ hbuf,
                                float* __restrict__ cbuf) {
    int i = blockIdx.x * blockDim.x + threadIdx.x;   // 0 .. 65535
    hbuf[i]                     = (__bf16)0.0f;
    hbuf[BATCH * HIDDEN + i]    = (__bf16)0.0f;
    cbuf[i]                     = 0.0f;
}

// ---------------------------------------------------------------------------
// Kernel: x_gates GEMM.  C[m][n] = sum_k x[m][k]*Wx[n][k] + bx[n], bf16 out.
// Block: 256 threads (8 waves).  Macro tile 128(M) x 64(N).
// Wave w owns M-tile m0 = blk.x*128 + w*16 and all four 16-col N-tiles,
// sharing a per-k-chunk Wx tile staged (async) in LDS by the whole block.
// ---------------------------------------------------------------------------
__global__ __launch_bounds__(256)
void lstm_gemm_xgates(const float* __restrict__ x,       // [S*B][INPUT]
                      const __bf16* __restrict__ Wx_bf,  // [GATES][INPUT]
                      const float* __restrict__ bx,      // [GATES]
                      __bf16* __restrict__ xg) {         // [S*B][GATES]
    __shared__ __align__(32) __bf16 Bsh[64 * 32];        // 4 KB Wx chunk

    const int tid  = threadIdx.x;
    const int lane = tid & 31;
    const int wave = tid >> 5;                  // 0..7
    const int m0   = blockIdx.x * 128 + wave * 16;
    const int n0   = blockIdx.y * 64;

    v8f acc[4] = {v8f{}, v8f{}, v8f{}, v8f{}};

    for (int kk = 0; kk < INPUT; kk += 32) {
        // Stage B: Bsh[j][k] = Wx_bf[n0+j][kk+k]   (j=0..63, k=0..31)
        {
            int j  = tid >> 2;
            int ko = (tid & 3) * 8;
            copy16_g2l(&Wx_bf[(size_t)(n0 + j) * INPUT + kk + ko],
                       &Bsh[j * 32 + ko]);
        }

        // A fragment straight from global f32 x, converted to bf16
        // (overlaps with the async B staging).
        v16bf a;
        {
            int row  = m0 + (lane & 15);
            int koff = (lane >> 4) ? 8 : 0;
            const float* p = x + (size_t)row * INPUT + kk + koff;
            v8f lo = *(const v8f*)(p);
            v8f hi = *(const v8f*)(p + 16);
#pragma unroll
            for (int i = 0; i < 8; ++i) {
                a[i]     = (__bf16)lo[i];
                a[8 + i] = (__bf16)hi[i];
            }
        }

        copy_fence_pre_barrier();
        __syncthreads();

        // Hoist all four B fragments, then issue the WMMAs back-to-back.
        v16bf b0 = load_b_frag_lds(&Bsh[0 * 16 * 32], lane);
        v16bf b1 = load_b_frag_lds(&Bsh[1 * 16 * 32], lane);
        v16bf b2 = load_b_frag_lds(&Bsh[2 * 16 * 32], lane);
        v16bf b3 = load_b_frag_lds(&Bsh[3 * 16 * 32], lane);
        acc[0] = wmma_bf16(a, b0, acc[0]);
        acc[1] = wmma_bf16(a, b1, acc[1]);
        acc[2] = wmma_bf16(a, b2, acc[2]);
        acc[3] = wmma_bf16(a, b3, acc[3]);
        __syncthreads();
    }

    // Epilogue: + bx, convert to bf16, scatter (8 rows x 1 col per lane).
    const int col = lane & 15;
    const int rsh = (lane >> 4) ? 8 : 0;
#pragma unroll
    for (int nt = 0; nt < 4; ++nt) {
        int   n    = n0 + nt * 16 + col;
        float bias = bx[n];
#pragma unroll
        for (int r = 0; r < 8; ++r) {
            int m = m0 + r + rsh;
            xg[(size_t)m * GATES + n] = (__bf16)(acc[nt][r] + bias);
        }
    }
}

// ---------------------------------------------------------------------------
// Kernel: one LSTM time step.  Grid: 64 blocks (one per 16 hidden cols),
// 256 threads = 8 waves.  Wave w: M-tile mt = w&3 (batch rows), gate pair
// gp = w>>2 (gates {0,1} or {2,3}).  Per k-chunk: h staged via async LDS
// copies, Wh tile staged via one TDM tensor_load_to_lds (4-way B reuse).
// Gate tiles meet in LDS for the fused cell update.
// ---------------------------------------------------------------------------
__global__ __launch_bounds__(256)
void lstm_step(const __bf16* __restrict__ Wh_bf,   // [GATES][HIDDEN]
               const __bf16* __restrict__ xg,      // [S*B][GATES]
               const float*  __restrict__ bh,      // [GATES]
               const __bf16* __restrict__ h_in,    // [BATCH][HIDDEN]
               __bf16*       __restrict__ h_out,   // [BATCH][HIDDEN]
               float*        __restrict__ c_buf,   // [BATCH][HIDDEN]
               float*        __restrict__ out,     // outputs|h_T|c_T
               int t) {
    __shared__ __align__(32) __bf16 hsh[64 * 32];       // 4 KB h chunk
#if USE_TDM
    __shared__ __align__(32) __bf16 Whsh[64 * 32];      // 4 KB Wh chunk
#endif
    __shared__ __align__(32) float  gsh[4 * 64 * 16];   // 16 KB gate tiles

    const int tid  = threadIdx.x;
    const int lane = tid & 31;
    const int wave = tid >> 5;
    const int mt   = wave & 3;       // batch-row tile
    const int gp   = wave >> 2;      // gate pair
    const int jblk = blockIdx.x;     // hidden column block (16 cols)
    const int g0   = 2 * gp;

    v8f acc0 = v8f{};
    v8f acc1 = v8f{};

    for (int kk = 0; kk < HIDDEN; kk += 32) {
#if USE_TDM
        if (wave == 0) tdm_load_wh(Wh_bf, Whsh, jblk, kk);
#endif
        // Stage h chunk: hsh[b][k] = h_in[b][kk+k]
        {
            int b  = tid >> 2;
            int ko = (tid & 3) * 8;
            copy16_g2l(&h_in[(size_t)b * HIDDEN + kk + ko],
                       &hsh[b * 32 + ko]);
        }
        copy_fence_pre_barrier();
#if USE_TDM
        if (wave == 0) __builtin_amdgcn_s_wait_tensorcnt(0);
#endif
        __syncthreads();

        v16bf a = load_a_frag_lds(&hsh[mt * 16 * 32], lane);
#if USE_TDM
        v16bf b0 = load_b_frag_lds(&Whsh[(g0 + 0) * 16 * 32], lane);
        v16bf b1 = load_b_frag_lds(&Whsh[(g0 + 1) * 16 * 32], lane);
#else
        v16bf b0 = load_b_frag_global(Wh_bf, (g0 + 0) * HIDDEN + jblk * 16,
                                      kk, HIDDEN, lane);
        v16bf b1 = load_b_frag_global(Wh_bf, (g0 + 1) * HIDDEN + jblk * 16,
                                      kk, HIDDEN, lane);
#endif
        acc0 = wmma_bf16(a, b0, acc0);
        acc1 = wmma_bf16(a, b1, acc1);
        __syncthreads();
    }

    // Park raw gate accumulators in LDS: gsh[gate][b][col]
    {
        const int col = lane & 15;
        const int rsh = (lane >> 4) ? 8 : 0;
#pragma unroll
        for (int r = 0; r < 8; ++r) {
            int brow = mt * 16 + r + rsh;
            gsh[((g0 + 0) * 64 + brow) * 16 + col] = acc0[r];
            gsh[((g0 + 1) * 64 + brow) * 16 + col] = acc1[r];
        }
    }
    __syncthreads();

    // Fused cell update: 1024 (b, col) elements, 4 per thread.
    const size_t xrow = (size_t)t * BATCH;
    for (int e = tid; e < 64 * 16; e += 256) {
        int b    = e >> 4;
        int cc   = e & 15;
        int hcol = jblk * 16 + cc;
        const __bf16* xgr = xg + (xrow + b) * GATES;

        float gi = gsh[(0 * 64 + b) * 16 + cc] + (float)xgr[0 * HIDDEN + hcol] + bh[0 * HIDDEN + hcol];
        float gf = gsh[(1 * 64 + b) * 16 + cc] + (float)xgr[1 * HIDDEN + hcol] + bh[1 * HIDDEN + hcol];
        float gg = gsh[(2 * 64 + b) * 16 + cc] + (float)xgr[2 * HIDDEN + hcol] + bh[2 * HIDDEN + hcol];
        float go = gsh[(3 * 64 + b) * 16 + cc] + (float)xgr[3 * HIDDEN + hcol] + bh[3 * HIDDEN + hcol];

        float i_t = fsigmoid(gi);
        float f_t = fsigmoid(gf);
        float g_t = tanhf(gg);
        float o_t = fsigmoid(go);

        size_t sidx = (size_t)b * HIDDEN + hcol;
        float  c_n  = f_t * c_buf[sidx] + i_t * g_t;
        float  h_n  = o_t * tanhf(c_n);

        c_buf[sidx] = c_n;
        h_out[sidx] = (__bf16)h_n;
        out[((size_t)b * SEQ + t) * HIDDEN + hcol] = h_n;
        if (t == SEQ - 1) {
            size_t base = (size_t)BATCH * SEQ * HIDDEN;
            out[base + sidx]                  = h_n;   // h_T
            out[base + BATCH * HIDDEN + sidx] = c_n;   // c_T
        }
    }
}

// ---------------------------------------------------------------------------
// Host launcher.
// Inputs: x [S,B,I] f32, Wx [4H,I] f32, bx [4H] f32, Wh [4H,H] f32, bh [4H].
// Output: outputs [B,S,H] f32, then h_T [B,H], then c_T [B,H].
// ---------------------------------------------------------------------------
extern "C" void kernel_launch(void* const* d_in, const int* in_sizes, int n_in,
                              void* d_out, int out_size, void* d_ws, size_t ws_size,
                              hipStream_t stream) {
    const float* x  = (const float*)d_in[0];
    const float* Wx = (const float*)d_in[1];
    const float* bx = (const float*)d_in[2];
    const float* Wh = (const float*)d_in[3];
    const float* bh = (const float*)d_in[4];
    float* out = (float*)d_out;

    // Workspace carve-up (all offsets >= 32B aligned).
    char* p = (char*)d_ws;
    __bf16* Wx_bf = (__bf16*)p;  p += (size_t)GATES * INPUT * sizeof(__bf16);   //   8 MB
    __bf16* Wh_bf = (__bf16*)p;  p += (size_t)GATES * HIDDEN * sizeof(__bf16);  //   8 MB
    __bf16* xg    = (__bf16*)p;  p += (size_t)SEQ * BATCH * GATES * sizeof(__bf16); // 256 MB
    __bf16* hbuf  = (__bf16*)p;  p += (size_t)2 * BATCH * HIDDEN * sizeof(__bf16);  // 256 KB
    float*  cbuf  = (float*)p;   p += (size_t)BATCH * HIDDEN * sizeof(float);       // 256 KB
    (void)ws_size; (void)in_sizes; (void)n_in; (void)out_size;

    // 1) Weights -> bf16 (stays L2-resident: 16 MB total in a 192 MB L2).
    {
        const size_t total = (size_t)2 * GATES * INPUT;       // 8M elements
        dim3 grid((unsigned)((total + 255) / 256));
        lstm_convert_weights<<<grid, dim3(256), 0, stream>>>(Wx, Wh, Wx_bf, Wh_bf);
    }

    // 2) Zero h0/c0 (re-done every call: deterministic under graph replay).
    lstm_init_state<<<dim3(256), dim3(256), 0, stream>>>(hbuf, cbuf);

    // 3) Big parallel projection: [32768 x 4096 x 1024] bf16 WMMA GEMM.
    lstm_gemm_xgates<<<dim3(256, 64), dim3(256), 0, stream>>>(x, Wx_bf, bx, xg);

    // 4) Sequential recurrence: 512 step kernels on one stream.
    for (int t = 0; t < SEQ; ++t) {
        const __bf16* h_in  = hbuf + (size_t)(t & 1) * BATCH * HIDDEN;
        __bf16*       h_out = hbuf + (size_t)((t + 1) & 1) * BATCH * HIDDEN;
        lstm_step<<<dim3(64), dim3(256), 0, stream>>>(Wh_bf, xg, bh, h_in, h_out,
                                                      cbuf, out, t);
    }
}
